// AnnoCluster_21638045237477
// MI455X (gfx1250) — matrix operand order repaired and checked by reading, not compiled, not run on hardware
//
#include <hip/hip_runtime.h>
#include <hip/hip_bf16.h>
#include <math.h>

// ---------------------------------------------------------------------------
// AnnoCluster forward on MI455X (gfx1250, wave32, WMMA).
// Round 3: weights pre-packed to bf16 in WMMA B-fragment order -> B fragment
// loads become 2x global_load_b128, no cvt, no strided dword loads.
// ---------------------------------------------------------------------------

typedef __bf16  v16bf __attribute__((ext_vector_type(16)));
typedef float   v8f   __attribute__((ext_vector_type(8)));

#define B_ROWS 4096
#define D_DIM  10000
#define H_DIM  128
#define Z_DIM  32
#define K_CLU  16

#define W1_TILES  313                         // ceil(10000/32)
#define W2_TILES  4                           // 128/32
#define W1P_ELEMS ((size_t)W1_TILES * H_DIM * 32)   // 1,282,048 bf16
#define W2P_ELEMS ((size_t)W2_TILES * D_DIM * 32)   // 1,280,000 bf16

// Packed B layout: element index = (((tile * N) + n) * 2 + half) * 16 + e
// so each lane's 16 bf16 fragment elements are 32 contiguous bytes.

// ---- one-shot weight packer: f32 [K x N] -> bf16 fragment order ------------
__global__ void __launch_bounds__(256)
k_pack_b(const float* __restrict__ W, __bf16* __restrict__ Wp,
         int Kdim, int N, int ntiles) {
    int idx = blockIdx.x * blockDim.x + threadIdx.x;  // (tile*N + n)*2 + half
    int total = ntiles * N * 2;
    if (idx >= total) return;
    int half = idx & 1;
    int t    = idx >> 1;
    int n    = t % N;
    int tile = t / N;
    int kbase = tile * 32 + half * 8;
    __bf16* dst = Wp + (size_t)idx * 16;
#pragma unroll
    for (int e = 0; e < 16; ++e) {
        int k = kbase + (e < 8 ? e : e + 8);
        float v = (k < Kdim) ? W[(size_t)k * N + n] : 0.0f;
        dst[e] = (__bf16)v;
    }
}

// ---- A fragment loaders (f32 memory -> bf16), per ISA lane layout ----------
__device__ __forceinline__ v16bf load_a(const float* __restrict__ arow,
                                        int k0, int half) {
    const float* p0 = arow + k0 + half * 8;
    const float* p1 = p0 + 16;
    v16bf a;
#pragma unroll
    for (int e = 0; e < 8; ++e) a[e] = (__bf16)p0[e];
#pragma unroll
    for (int e = 0; e < 8; ++e) a[8 + e] = (__bf16)p1[e];
    return a;
}

// Tail (16 valid K): elements 0..7 valid, 8..15 zero — no compares needed.
__device__ __forceinline__ v16bf load_a_tail16(const float* __restrict__ arow,
                                               int k0, int half) {
    const float* p0 = arow + k0 + half * 8;
    v16bf a;
#pragma unroll
    for (int e = 0; e < 8; ++e) a[e] = (__bf16)p0[e];
#pragma unroll
    for (int e = 0; e < 8; ++e) a[8 + e] = (__bf16)0.0f;
    return a;
}

#define WMMA_BF16(A, Bf, C) \
    __builtin_amdgcn_wmma_f32_16x16x32_bf16(false, (A), false, (Bf), (short)0, (C), false, false)

// ---- Kernel 1: h1 = relu(x @ enc_w1 + enc_b1)   [4096 x 128] ---------------
// 8 waves, 2(M) x 4(N); wave computes 16x32 (2 acc). Block: 32 x 128.
__global__ void __launch_bounds__(256)
k_gemm1_relu(const float* __restrict__ X, const __bf16* __restrict__ W1p,
             const float* __restrict__ bias, float* __restrict__ H1) {
    int wave = threadIdx.x >> 5;
    int lane = threadIdx.x & 31;
    int half = lane >> 4;

    int m0 = blockIdx.y * 32 + (wave >> 2) * 16;
    int n0 = (wave & 3) * 32;

    const float* arow = X + (size_t)(m0 + (lane & 15)) * D_DIM;
    const __bf16* bp0 = W1p + (((size_t)(n0 + (lane & 15))) * 2 + half) * 16;
    const __bf16* bp1 = bp0 + 16 * 32;              // +16 columns (32 elems/col)
    const size_t tstride = (size_t)H_DIM * 32;      // packed elems per K-tile

    v8f acc0 = {}, acc1 = {};

    for (int t = 0; t < W1_TILES - 1; ++t) {
        v16bf a  = load_a(arow, t * 32, half);
        v16bf b0 = *(const v16bf*)(bp0 + (size_t)t * tstride);
        v16bf b1 = *(const v16bf*)(bp1 + (size_t)t * tstride);
        acc0 = WMMA_BF16(a, b0, acc0);
        acc1 = WMMA_BF16(a, b1, acc1);
    }
    {   // K tail 9984..9999: packed B already zero-filled beyond K
        constexpr int t = W1_TILES - 1;
        v16bf a  = load_a_tail16(arow, t * 32, half);
        v16bf b0 = *(const v16bf*)(bp0 + (size_t)t * tstride);
        v16bf b1 = *(const v16bf*)(bp1 + (size_t)t * tstride);
        acc0 = WMMA_BF16(a, b0, acc0);
        acc1 = WMMA_BF16(a, b1, acc1);
    }

    int n     = n0 + (lane & 15);
    int mbase = m0 + half * 8;
    float bv0 = bias[n];
    float bv1 = bias[n + 16];
#pragma unroll
    for (int r = 0; r < 8; ++r) {
        float v0 = acc0[r] + bv0;
        float v1 = acc1[r] + bv1;
        size_t rowoff = (size_t)(mbase + r) * H_DIM;
        H1[rowoff + n]      = v0 > 0.0f ? v0 : 0.0f;
        H1[rowoff + n + 16] = v1 > 0.0f ? v1 : 0.0f;
    }
}

// ---- Kernel 2: per-row encoder head + clustering ---------------------------
__global__ void __launch_bounds__(32)
k_cluster(const float* __restrict__ H1, const float* __restrict__ W2,
          const float* __restrict__ b2, const float* __restrict__ emb,
          float* __restrict__ ze_out, float* __restrict__ zq_out,
          float* __restrict__ k_out, float* __restrict__ zdist_out,
          float* __restrict__ prob_out,
          float* __restrict__ ze_ws, float* __restrict__ zq_ws) {
    __shared__ float sh[H_DIM];
    __shared__ float zsh[Z_DIM];
    __shared__ float dsh[K_CLU];
    __shared__ float psh[K_CLU];
    __shared__ float ssum;
    __shared__ int   sbest;

    int row  = blockIdx.x;
    int lane = threadIdx.x;

    for (int j = lane; j < H_DIM; j += 32)
        sh[j] = H1[(size_t)row * H_DIM + j];
    __syncthreads();

    float acc = b2[lane];
#pragma unroll 8
    for (int j = 0; j < H_DIM; ++j)
        acc += sh[j] * W2[j * Z_DIM + lane];
    zsh[lane] = acc;
    ze_out[(size_t)row * Z_DIM + lane] = acc;
    ze_ws [(size_t)row * Z_DIM + lane] = acc;
    __syncthreads();

    if (lane < K_CLU) {
        float d = 0.0f;
#pragma unroll
        for (int z = 0; z < Z_DIM; ++z) {
            float t = zsh[z] - emb[lane * Z_DIM + z];
            d += t * t;
        }
        dsh[lane] = d;
        zdist_out[(size_t)row * K_CLU + lane] = d;
        psh[lane] = powf(1.0f + d * 0.1f, -5.5f);   // (1+d/T)^(-(T+1)/2), T=10
    }
    __syncthreads();

    if (lane == 0) {
        float s = 0.0f;
        int best = 0;
        float bd = dsh[0];
        for (int i = 0; i < K_CLU; ++i) {
            s += psh[i];
            if (dsh[i] < bd) { bd = dsh[i]; best = i; }  // argmax prob == argmin dist
        }
        ssum = s;
        sbest = best;
        k_out[row] = (float)best;
    }
    __syncthreads();

    if (lane < K_CLU)
        prob_out[(size_t)row * K_CLU + lane] = psh[lane] / ssum;

    float zq = emb[sbest * Z_DIM + lane];
    zq_out[(size_t)row * Z_DIM + lane] = zq;
    zq_ws [(size_t)row * Z_DIM + lane] = zq;
}

// ---- Kernel 3: decoder hidden layers (both e and q), Z=32 MACs each --------
__global__ void __launch_bounds__(256)
k_dec_hidden(const float* __restrict__ ze, const float* __restrict__ zq,
             const float* __restrict__ we1, const float* __restrict__ be1,
             const float* __restrict__ wq1, const float* __restrict__ bq1,
             float* __restrict__ he, float* __restrict__ hq) {
    int idx = blockIdx.x * blockDim.x + threadIdx.x;   // 0 .. B*H-1
    int row = idx >> 7;
    int h   = idx & (H_DIM - 1);
    const float* zer = ze + (size_t)row * Z_DIM;
    const float* zqr = zq + (size_t)row * Z_DIM;
    float ae = be1[h];
    float aq = bq1[h];
#pragma unroll
    for (int z = 0; z < Z_DIM; ++z) {
        ae = fmaf(zer[z], we1[z * H_DIM + h], ae);
        aq = fmaf(zqr[z], wq1[z * H_DIM + h], aq);
    }
    he[idx] = ae > 0.0f ? ae : 0.0f;
    hq[idx] = aq > 0.0f ? aq : 0.0f;
}

// ---- Kernel 4: x = h @ dec_w2 + b2   [4096 x 10000], e/q via grid.z --------
// K=128: 4 fully unrolled WMMA steps. Loads clamp n (packed array in-bounds);
// only stores are guarded -> branch-free inner loop.
__global__ void __launch_bounds__(256)
k_gemm2(const float* __restrict__ He, const float* __restrict__ Hq,
        const __bf16* __restrict__ Wep, const float* __restrict__ be,
        const __bf16* __restrict__ Wqp, const float* __restrict__ bq,
        float* __restrict__ Xe, float* __restrict__ Xq) {
    const float* Hh; const __bf16* Wp; const float* bias; float* Out;
    if (blockIdx.z == 0) { Hh = He; Wp = Wep; bias = be; Out = Xe; }
    else                 { Hh = Hq; Wp = Wqp; bias = bq; Out = Xq; }

    int wave = threadIdx.x >> 5;
    int lane = threadIdx.x & 31;
    int half = lane >> 4;

    int m0 = blockIdx.y * 32 + (wave >> 2) * 16;
    int n0 = blockIdx.x * 128 + (wave & 3) * 32;
    int na = n0 + (lane & 15);
    int nb = na + 16;
    int nca = na < D_DIM ? na : D_DIM - 1;          // clamp for loads
    int ncb = nb < D_DIM ? nb : D_DIM - 1;

    const float* arow = Hh + (size_t)(m0 + (lane & 15)) * H_DIM;
    const __bf16* bpa = Wp + (((size_t)nca) * 2 + half) * 16;
    const __bf16* bpb = Wp + (((size_t)ncb) * 2 + half) * 16;
    const size_t tstride = (size_t)D_DIM * 32;      // packed elems per K-tile

    v8f acc0 = {}, acc1 = {};
#pragma unroll
    for (int t = 0; t < W2_TILES; ++t) {
        v16bf a  = load_a(arow, t * 32, half);
        v16bf b0 = *(const v16bf*)(bpa + (size_t)t * tstride);
        v16bf b1 = *(const v16bf*)(bpb + (size_t)t * tstride);
        acc0 = WMMA_BF16(a, b0, acc0);
        acc1 = WMMA_BF16(a, b1, acc1);
    }

    int mbase = m0 + half * 8;
    if (na < D_DIM) {
        float bv = bias[na];
#pragma unroll
        for (int r = 0; r < 8; ++r)
            Out[(size_t)(mbase + r) * D_DIM + na] = acc0[r] + bv;
    }
    if (nb < D_DIM) {
        float bv = bias[nb];
#pragma unroll
        for (int r = 0; r < 8; ++r)
            Out[(size_t)(mbase + r) * D_DIM + nb] = acc1[r] + bv;
    }
}

// ---------------------------------------------------------------------------

extern "C" void kernel_launch(void* const* d_in, const int* in_sizes, int n_in,
                              void* d_out, int out_size, void* d_ws, size_t ws_size,
                              hipStream_t stream) {
    const float* x        = (const float*)d_in[0];
    const float* enc_w1   = (const float*)d_in[1];
    const float* enc_b1   = (const float*)d_in[2];
    const float* enc_w2   = (const float*)d_in[3];
    const float* enc_b2   = (const float*)d_in[4];
    const float* emb      = (const float*)d_in[5];
    const float* dec_e_w1 = (const float*)d_in[6];
    const float* dec_e_b1 = (const float*)d_in[7];
    const float* dec_e_w2 = (const float*)d_in[8];
    const float* dec_e_b2 = (const float*)d_in[9];
    const float* dec_q_w1 = (const float*)d_in[10];
    const float* dec_q_b1 = (const float*)d_in[11];
    const float* dec_q_w2 = (const float*)d_in[12];
    const float* dec_q_b2 = (const float*)d_in[13];

    // Output tuple layout (flat, float32):
    float* out   = (float*)d_out;
    float* x_e   = out;                                       // [B,D]
    float* x_q   = x_e + (size_t)B_ROWS * D_DIM;              // [B,D]
    float* z_e   = x_q + (size_t)B_ROWS * D_DIM;              // [B,Z]
    float* z_q   = z_e + (size_t)B_ROWS * Z_DIM;              // [B,Z]
    float* k_o   = z_q + (size_t)B_ROWS * Z_DIM;              // [B]
    float* zdist = k_o + (size_t)B_ROWS;                      // [B,K]
    float* prob  = zdist + (size_t)B_ROWS * K_CLU;            // [B,K]

    // Workspace layout: h1 | ze | zq | he | hq | w1p | w2ep | w2qp
    float*  ws    = (float*)d_ws;
    float*  h1    = ws;                                       // [B,H]
    float*  ze_ws = h1 + (size_t)B_ROWS * H_DIM;              // [B,Z]
    float*  zq_ws = ze_ws + (size_t)B_ROWS * Z_DIM;           // [B,Z]
    float*  he_ws = zq_ws + (size_t)B_ROWS * Z_DIM;           // [B,H]
    float*  hq_ws = he_ws + (size_t)B_ROWS * H_DIM;           // [B,H]
    __bf16* w1p   = (__bf16*)(hq_ws + (size_t)B_ROWS * H_DIM);
    __bf16* w2ep  = w1p + W1P_ELEMS;                          // 32B-aligned
    __bf16* w2qp  = w2ep + W2P_ELEMS;

    // 0) pack weights to bf16 fragment order (one-shot, trivial cost)
    {
        int tot1 = W1_TILES * H_DIM * 2;                      // 80,128
        k_pack_b<<<(tot1 + 255) / 256, 256, 0, stream>>>(enc_w1, w1p,
                                                         D_DIM, H_DIM, W1_TILES);
        int tot2 = W2_TILES * D_DIM * 2;                      // 80,000
        k_pack_b<<<(tot2 + 255) / 256, 256, 0, stream>>>(dec_e_w2, w2ep,
                                                         H_DIM, D_DIM, W2_TILES);
        k_pack_b<<<(tot2 + 255) / 256, 256, 0, stream>>>(dec_q_w2, w2qp,
                                                         H_DIM, D_DIM, W2_TILES);
    }
    // 1) encoder GEMM1 + ReLU  (WMMA bf16)
    {
        dim3 grid(1, B_ROWS / 32, 1);                         // (1, 128)
        k_gemm1_relu<<<grid, 256, 0, stream>>>(x, w1p, enc_b1, h1);
    }
    // 2) encoder head + clustering (one wave per row)
    {
        k_cluster<<<B_ROWS, 32, 0, stream>>>(h1, enc_w2, enc_b2, emb,
                                             z_e, z_q, k_o, zdist, prob,
                                             ze_ws, zq_ws);
    }
    // 3) decoder hidden layers
    {
        int threads = B_ROWS * H_DIM;                         // 524288
        k_dec_hidden<<<threads / 256, 256, 0, stream>>>(
            ze_ws, zq_ws, dec_e_w1, dec_e_b1, dec_q_w1, dec_q_b1,
            he_ws, hq_ws);
    }
    // 4) decoder GEMM2 (e and q), WMMA bf16
    {
        dim3 grid((D_DIM + 127) / 128, B_ROWS / 32, 2);       // (79, 128, 2)
        k_gemm2<<<grid, 256, 0, stream>>>(he_ws, hq_ws,
                                          w2ep, dec_e_b2,
                                          w2qp, dec_q_b2,
                                          x_e, x_q);
    }
}
